// MultiheadAttention_63067299774603
// MI455X (gfx1250) — matrix operand order repaired and checked by reading, not compiled, not run on hardware
//
#include <hip/hip_runtime.h>

// ---------------------------------------------------------------------------
// MHA forward for MI455X (gfx1250, wave32, WMMA bf16 path), register-blocked.
// Model: B=2, S=2048, D=1024, H=16, depth=64
// Outputs: out [B,S,D] f32  ++  weights [B,H,S,S] f32
// ---------------------------------------------------------------------------

#define BATCH   2
#define SEQ     2048
#define DMODEL  1024
#define NHEADS  16
#define DEPTH   64
#define BH      (BATCH * NHEADS)        // 32
#define MTOT    (BATCH * SEQ)           // 4096

typedef __attribute__((ext_vector_type(16))) __bf16 v16bf;
typedef __attribute__((ext_vector_type(8)))  __bf16 v8bf;
typedef __attribute__((ext_vector_type(8)))  float  v8f;
typedef __attribute__((ext_vector_type(4)))  float  v4f;

union FragB16 { v16bf v; v8bf h[2]; };

// ---- fragment builders (layouts per CDNA5 ISA §7.12.2) ----------

__device__ inline v8bf cvt8_f32(const float* p) {
  v4f a = *(const v4f*)p;
  v4f b = *(const v4f*)(p + 4);
  v8bf r;
  r[0] = (__bf16)a[0]; r[1] = (__bf16)a[1]; r[2] = (__bf16)a[2]; r[3] = (__bf16)a[3];
  r[4] = (__bf16)b[0]; r[5] = (__bf16)b[1]; r[6] = (__bf16)b[2]; r[7] = (__bf16)b[3];
  return r;
}

// 16-bit A 16x32: lane<16 -> K runs {klo..klo+7, 16+klo..+7} klo=0; lane>=16 klo=8.
__device__ inline v16bf loadA_f32(const float* __restrict__ row, int kbase, int lane) {
  const int klo = (lane & 16) ? 8 : 0;
  FragB16 f;
  f.h[0] = cvt8_f32(row + kbase + klo);
  f.h[1] = cvt8_f32(row + kbase + 16 + klo);
  return f.v;
}

__device__ inline v16bf loadA_bf16(const __bf16* __restrict__ row, int kbase, int lane) {
  const int klo = (lane & 16) ? 8 : 0;
  FragB16 f;
  f.h[0] = *(const v8bf*)(row + kbase + klo);
  f.h[1] = *(const v8bf*)(row + kbase + 16 + klo);
  return f.v;
}

// 16-bit B 32x16: lane<16 -> K runs {klo..klo+7, 16+klo..+7} klo=0; lane>=16 klo=8; n = lane&15.
__device__ inline v16bf loadB_bf16(const __bf16* __restrict__ rown, int kbase, int lane) {
  const int klo = (lane & 16) ? 8 : 0;
  FragB16 f;
  f.h[0] = *(const v8bf*)(rown + kbase + klo);
  f.h[1] = *(const v8bf*)(rown + kbase + 16 + klo);
  return f.v;
}

__device__ inline v8f wmma_bf16(v16bf a, v16bf b, v8f c) {
  return __builtin_amdgcn_wmma_f32_16x16x32_bf16(
      false, a, false, b, (short)0, c, false, false);
}

// ---- K0: f32 -> bf16 weight conversion -------------------------------------
__global__ void cvt_bf16_kernel(const float* __restrict__ src,
                                __bf16* __restrict__ dst, int n) {
  int i = blockIdx.x * blockDim.x + threadIdx.x;
  if (i < n) dst[i] = (__bf16)src[i];
}

// ---- K1: input projection, y = x @ W^T, 16x64 tile / wave ------------------
// grid = (MTOT/16, DMODEL/64), block = 32. Output tile is one full head.
__global__ __launch_bounds__(32) void proj_qkv_kernel(
    const float*  __restrict__ X,    // [MTOT][DMODEL] f32
    const __bf16* __restrict__ Wb,   // [DMODEL][DMODEL] bf16 (row f, K-contig)
    __bf16* __restrict__ Out,        // head-split destination
    int vtranspose)                  // 0: [BH][S][64]   1: [BH][64][S]
{
  const int lane = threadIdx.x & 31;
  const int m0   = blockIdx.x * 16;
  const int f0   = blockIdx.y * 64;          // head-aligned
  const int col  = lane & 15;

  const float*  arow = X + (size_t)(m0 + col) * DMODEL;
  const __bf16* br0 = Wb + (size_t)(f0 +  0 + col) * DMODEL;
  const __bf16* br1 = Wb + (size_t)(f0 + 16 + col) * DMODEL;
  const __bf16* br2 = Wb + (size_t)(f0 + 32 + col) * DMODEL;
  const __bf16* br3 = Wb + (size_t)(f0 + 48 + col) * DMODEL;

  v8f acc0 = {}, acc1 = {}, acc2 = {}, acc3 = {};
  for (int kb = 0; kb < DMODEL; kb += 32) {
    __builtin_prefetch(arow + kb + 64, 0, 1);      // next k-blocks of A
    const v16bf a = loadA_f32(arow, kb, lane);     // reused by 4 WMMAs
    acc0 = wmma_bf16(a, loadB_bf16(br0, kb, lane), acc0);
    acc1 = wmma_bf16(a, loadB_bf16(br1, kb, lane), acc1);
    acc2 = wmma_bf16(a, loadB_bf16(br2, kb, lane), acc2);
    acc3 = wmma_bf16(a, loadB_bf16(br3, kb, lane), acc3);
  }

  const int mhi = (lane >> 4) * 8;
  const int h   = f0 >> 6;
#pragma unroll
  for (int r = 0; r < 8; ++r) {
    const int m  = m0 + mhi + r;
    const int b_ = m >> 11, s = m & (SEQ - 1);
    const size_t headbase = (size_t)(b_ * NHEADS + h);
    if (!vtranspose) {
      __bf16* orow = Out + (headbase * SEQ + s) * DEPTH;
      orow[ 0 + col] = (__bf16)acc0[r];
      orow[16 + col] = (__bf16)acc1[r];
      orow[32 + col] = (__bf16)acc2[r];
      orow[48 + col] = (__bf16)acc3[r];
    } else {
      __bf16* ocol = Out + headbase * DEPTH * SEQ + s;
      ocol[(size_t)( 0 + col) * SEQ] = (__bf16)acc0[r];
      ocol[(size_t)(16 + col) * SEQ] = (__bf16)acc1[r];
      ocol[(size_t)(32 + col) * SEQ] = (__bf16)acc2[r];
      ocol[(size_t)(48 + col) * SEQ] = (__bf16)acc3[r];
    }
  }
}

// ---- K2a: logits = (Qh Kh^T)/8 + mask, 32 q-rows / wave --------------------
// grid = (SEQ/32, BH), block = 32
__global__ __launch_bounds__(32) void logits_kernel(
    const __bf16* __restrict__ Qh,    // [BH][S][64]
    const __bf16* __restrict__ Kh,    // [BH][S][64]
    const float*  __restrict__ mask,  // [S][S]
    float* __restrict__ Wraw,         // [BH][S][S] raw masked scaled logits
    float* __restrict__ stats)        // [BH*S][2] = {rowmax, rowsumexp}
{
  const int lane = threadIdx.x & 31;
  const int q0   = blockIdx.x * 32;
  const int bh   = blockIdx.y;
  const int col  = lane & 15;
  const int mhi  = (lane >> 4) * 8;

  // Q fragments for two 16-row blocks, held in registers for the whole k-loop
  const __bf16* qrow0 = Qh + ((size_t)bh * SEQ + q0 + col) * DEPTH;
  const __bf16* qrow1 = qrow0 + 16 * DEPTH;
  const v16bf a00 = loadA_bf16(qrow0, 0, lane), a01 = loadA_bf16(qrow0, 32, lane);
  const v16bf a10 = loadA_bf16(qrow1, 0, lane), a11 = loadA_bf16(qrow1, 32, lane);

  float m[16], ssum[16];
#pragma unroll
  for (int r = 0; r < 16; ++r) { m[r] = -3.0e38f; ssum[r] = 0.0f; }

  const float scale = 0.125f;  // 1/sqrt(64)
  float* wbase = Wraw + (size_t)bh * SEQ * SEQ;

  for (int kt = 0; kt < SEQ / 16; ++kt) {
    const __bf16* krow = Kh + ((size_t)bh * SEQ + kt * 16 + col) * DEPTH;
    const v16bf b0 = loadB_bf16(krow, 0, lane);
    const v16bf b1 = loadB_bf16(krow, 32, lane);    // each feeds 2 WMMAs
    v8f accA = {}, accB = {};
    accA = wmma_bf16(a00, b0, accA);
    accB = wmma_bf16(a10, b0, accB);
    accA = wmma_bf16(a01, b1, accA);
    accB = wmma_bf16(a11, b1, accB);
    const int k = kt * 16 + col;
#pragma unroll
    for (int r = 0; r < 8; ++r) {
      {
        const int q = q0 + mhi + r;
        const float x = accA[r] * scale + mask[(size_t)q * SEQ + k];
        wbase[(size_t)q * SEQ + k] = x;
        const float mn = fmaxf(m[r], x);
        ssum[r] = ssum[r] * __expf(m[r] - mn) + __expf(x - mn);
        m[r] = mn;
      }
      {
        const int q = q0 + 16 + mhi + r;
        const float x = accB[r] * scale + mask[(size_t)q * SEQ + k];
        wbase[(size_t)q * SEQ + k] = x;
        const float mn = fmaxf(m[8 + r], x);
        ssum[8 + r] = ssum[8 + r] * __expf(m[8 + r] - mn) + __expf(x - mn);
        m[8 + r] = mn;
      }
    }
  }

  // cross-lane combine within each 16-lane half
#pragma unroll
  for (int r = 0; r < 16; ++r) {
#pragma unroll
    for (int off = 1; off < 16; off <<= 1) {
      const float mo = __shfl_xor(m[r],    off, 16);
      const float so = __shfl_xor(ssum[r], off, 16);
      const float mn = fmaxf(m[r], mo);
      ssum[r] = ssum[r] * __expf(m[r] - mn) + so * __expf(mo - mn);
      m[r] = mn;
    }
  }
  if (col == 0) {
#pragma unroll
    for (int r = 0; r < 8; ++r) {
      const size_t rowA = (size_t)bh * SEQ + q0 + mhi + r;
      const size_t rowB = rowA + 16;
      stats[rowA * 2 + 0] = m[r];      stats[rowA * 2 + 1] = ssum[r];
      stats[rowB * 2 + 0] = m[8 + r];  stats[rowB * 2 + 1] = ssum[8 + r];
    }
  }
}

// ---- K2b: normalize weights in place: w = exp(x - M) / S -------------------
__global__ __launch_bounds__(256) void norm_kernel(
    float* __restrict__ W, const float* __restrict__ stats)
{
  const size_t i = (size_t)blockIdx.x * blockDim.x + threadIdx.x;  // float4 idx
  const size_t idx4 = i * 4;
  const size_t row  = idx4 >> 11;  // / SEQ
  const float M   = stats[row * 2 + 0];
  const float inv = 1.0f / stats[row * 2 + 1];
  v4f x = *(v4f*)(W + idx4);
  x[0] = __expf(x[0] - M) * inv;
  x[1] = __expf(x[1] - M) * inv;
  x[2] = __expf(x[2] - M) * inv;
  x[3] = __expf(x[3] - M) * inv;
  *(v4f*)(W + idx4) = x;
}

// ---- K3: attn = P @ Vh, 32 q-rows x 64 depth / wave ------------------------
// grid = (SEQ/32, BH), block = 32
__global__ __launch_bounds__(32) void attnv_kernel(
    const float*  __restrict__ P,    // [BH][S][S] normalized weights f32
    const __bf16* __restrict__ VhT,  // [BH][64][S]
    __bf16* __restrict__ Cc)         // concat [B][S][DMODEL] bf16
{
  const int lane = threadIdx.x & 31;
  const int q0   = blockIdx.x * 32;
  const int bh   = blockIdx.y;
  const int col  = lane & 15;

  const float* prow0 = P + ((size_t)bh * SEQ + q0 + col) * SEQ;
  const float* prow1 = prow0 + (size_t)16 * SEQ;
  const __bf16* vb    = VhT + (size_t)bh * DEPTH * SEQ;
  const __bf16* vrow0 = vb + (size_t)( 0 + col) * SEQ;
  const __bf16* vrow1 = vb + (size_t)(16 + col) * SEQ;
  const __bf16* vrow2 = vb + (size_t)(32 + col) * SEQ;
  const __bf16* vrow3 = vb + (size_t)(48 + col) * SEQ;

  v8f ac00 = {}, ac01 = {}, ac02 = {}, ac03 = {};
  v8f ac10 = {}, ac11 = {}, ac12 = {}, ac13 = {};
  for (int kb = 0; kb < SEQ; kb += 32) {
    __builtin_prefetch(prow0 + kb + 64, 0, 1);
    __builtin_prefetch(prow1 + kb + 64, 0, 1);
    const v16bf aA = loadA_f32(prow0, kb, lane);
    const v16bf aB = loadA_f32(prow1, kb, lane);
    const v16bf b0 = loadB_bf16(vrow0, kb, lane);   // each feeds 2 WMMAs
    const v16bf b1 = loadB_bf16(vrow1, kb, lane);
    const v16bf b2 = loadB_bf16(vrow2, kb, lane);
    const v16bf b3 = loadB_bf16(vrow3, kb, lane);
    ac00 = wmma_bf16(aA, b0, ac00);  ac10 = wmma_bf16(aB, b0, ac10);
    ac01 = wmma_bf16(aA, b1, ac01);  ac11 = wmma_bf16(aB, b1, ac11);
    ac02 = wmma_bf16(aA, b2, ac02);  ac12 = wmma_bf16(aB, b2, ac12);
    ac03 = wmma_bf16(aA, b3, ac03);  ac13 = wmma_bf16(aB, b3, ac13);
  }

  const int mhi = (lane >> 4) * 8;
  const int b_  = bh >> 4, h = bh & (NHEADS - 1);
#pragma unroll
  for (int r = 0; r < 8; ++r) {
    {
      const int q = q0 + mhi + r;
      __bf16* orow = Cc + ((size_t)b_ * SEQ + q) * DMODEL + h * DEPTH;
      orow[ 0 + col] = (__bf16)ac00[r];
      orow[16 + col] = (__bf16)ac01[r];
      orow[32 + col] = (__bf16)ac02[r];
      orow[48 + col] = (__bf16)ac03[r];
    }
    {
      const int q = q0 + 16 + mhi + r;
      __bf16* orow = Cc + ((size_t)b_ * SEQ + q) * DMODEL + h * DEPTH;
      orow[ 0 + col] = (__bf16)ac10[r];
      orow[16 + col] = (__bf16)ac11[r];
      orow[32 + col] = (__bf16)ac12[r];
      orow[48 + col] = (__bf16)ac13[r];
    }
  }
}

// ---- K4: out = concat @ wfc^T, 16x64 tile / wave, f32 store ----------------
// grid = (MTOT/16, DMODEL/64), block = 32
__global__ __launch_bounds__(32) void proj_out_kernel(
    const __bf16* __restrict__ Cc,   // [MTOT][DMODEL] bf16
    const __bf16* __restrict__ Wb,   // [DMODEL][DMODEL] bf16
    float* __restrict__ Out)         // [MTOT][DMODEL] f32
{
  const int lane = threadIdx.x & 31;
  const int m0   = blockIdx.x * 16;
  const int f0   = blockIdx.y * 64;
  const int col  = lane & 15;

  const __bf16* arow = Cc + (size_t)(m0 + col) * DMODEL;
  const __bf16* br0 = Wb + (size_t)(f0 +  0 + col) * DMODEL;
  const __bf16* br1 = Wb + (size_t)(f0 + 16 + col) * DMODEL;
  const __bf16* br2 = Wb + (size_t)(f0 + 32 + col) * DMODEL;
  const __bf16* br3 = Wb + (size_t)(f0 + 48 + col) * DMODEL;

  v8f acc0 = {}, acc1 = {}, acc2 = {}, acc3 = {};
  for (int kb = 0; kb < DMODEL; kb += 32) {
    const v16bf a = loadA_bf16(arow, kb, lane);
    acc0 = wmma_bf16(a, loadB_bf16(br0, kb, lane), acc0);
    acc1 = wmma_bf16(a, loadB_bf16(br1, kb, lane), acc1);
    acc2 = wmma_bf16(a, loadB_bf16(br2, kb, lane), acc2);
    acc3 = wmma_bf16(a, loadB_bf16(br3, kb, lane), acc3);
  }

  const int mhi = (lane >> 4) * 8;
#pragma unroll
  for (int r = 0; r < 8; ++r) {
    const int m = m0 + mhi + r;
    float* orow = Out + (size_t)m * DMODEL + f0;
    orow[ 0 + col] = acc0[r];
    orow[16 + col] = acc1[r];
    orow[32 + col] = acc2[r];
    orow[48 + col] = acc3[r];
  }
}

// ---------------------------------------------------------------------------
extern "C" void kernel_launch(void* const* d_in, const int* in_sizes, int n_in,
                              void* d_out, int out_size, void* d_ws, size_t ws_size,
                              hipStream_t stream) {
  (void)in_sizes; (void)n_in; (void)out_size; (void)ws_size;

  const float* q    = (const float*)d_in[0];
  const float* k    = (const float*)d_in[1];
  const float* v    = (const float*)d_in[2];
  const float* mask = (const float*)d_in[3];
  const float* wq   = (const float*)d_in[4];
  const float* wk   = (const float*)d_in[5];
  const float* wv   = (const float*)d_in[6];
  const float* wfc  = (const float*)d_in[7];

  float* out_p = (float*)d_out;                         // [B,S,D]
  float* w_p   = out_p + (size_t)MTOT * DMODEL;         // [B,H,S,S]

  // workspace layout
  char* ws = (char*)d_ws;
  __bf16* Qh    = (__bf16*)(ws + 0);              //  8 MiB  [BH][S][64]
  __bf16* Kh    = (__bf16*)(ws + (8u  << 20));    //  8 MiB  [BH][S][64]
  __bf16* VhT   = (__bf16*)(ws + (16u << 20));    //  8 MiB  [BH][64][S]
  __bf16* Cc    = (__bf16*)(ws + (24u << 20));    //  8 MiB  [B][S][D]
  __bf16* wqb   = (__bf16*)(ws + (32u << 20));    //  2 MiB each
  __bf16* wkb   = (__bf16*)(ws + (34u << 20));
  __bf16* wvb   = (__bf16*)(ws + (36u << 20));
  __bf16* wfcb  = (__bf16*)(ws + (38u << 20));
  float*  stats = (float*)  (ws + (40u << 20));   //  512 KiB  [BH*S][2]

  const int NW = DMODEL * DMODEL;

  // K0: weight conversion to bf16
  cvt_bf16_kernel<<<NW / 256, 256, 0, stream>>>(wq,  wqb,  NW);
  cvt_bf16_kernel<<<NW / 256, 256, 0, stream>>>(wk,  wkb,  NW);
  cvt_bf16_kernel<<<NW / 256, 256, 0, stream>>>(wv,  wvb,  NW);
  cvt_bf16_kernel<<<NW / 256, 256, 0, stream>>>(wfc, wfcb, NW);

  // K1: input projections (V stored head-transposed for the P@V B-fragments)
  dim3 gProj(MTOT / 16, DMODEL / DEPTH);  // (256, 16)
  proj_qkv_kernel<<<gProj, 32, 0, stream>>>(q, wqb, Qh,  0);
  proj_qkv_kernel<<<gProj, 32, 0, stream>>>(k, wkb, Kh,  0);
  proj_qkv_kernel<<<gProj, 32, 0, stream>>>(v, wvb, VhT, 1);

  // K2a: masked scaled logits + online softmax stats
  dim3 gAttn(SEQ / 32, BH);               // (64, 32)
  logits_kernel<<<gAttn, 32, 0, stream>>>(Qh, Kh, mask, w_p, stats);

  // K2b: normalize weights in place (final `weights` output)
  {
    const size_t n4 = (size_t)BH * SEQ * SEQ / 4;   // 33,554,432
    norm_kernel<<<(unsigned)(n4 / 256), 256, 0, stream>>>(w_p, stats);
  }

  // K3: attn = P @ V  -> bf16 merged-head concat
  attnv_kernel<<<gAttn, 32, 0, stream>>>(w_p, VhT, Cc);

  // K4: output projection -> f32 `out`
  proj_out_kernel<<<gProj, 32, 0, stream>>>(Cc, wfcb, out_p);
}
